// RPNHead_15642270892527
// MI455X (gfx1250) — matrix operand order repaired and checked by reading, not compiled, not run on hardware
//
#include <hip/hip_runtime.h>

typedef __bf16 bf16_t;
typedef __attribute__((ext_vector_type(16))) __bf16 v16bf;
typedef __attribute__((ext_vector_type(8)))  __bf16 v8bf;
typedef __attribute__((ext_vector_type(8)))  float  v8f;

#define B_IMG  16
#define H_IMG  37
#define W_IMG  37
#define CIN    1024
#define CMID   512
#define COUT   120
#define NPIX   (H_IMG * W_IMG)        // 1369
#define MTOT   (B_IMG * NPIX)         // 21904 (divisible by 16)
#define MTILES (MTOT / 16)            // 1369
#define MBLK2  ((MTILES + 1) / 2)     // 685: conv1x1 blocks (2 M-tiles)
#define MBLK4  ((MTILES + 3) / 4)     // 343: conv3x3 blocks (4 M-tiles, 64 pixels)

// ---- workspace layout (bytes) ----
#define WS_FMAPB 0                                   // NHWC bf16 fmap: 22,429,696 elems
#define WS_W1P   (44859392)                          // packed W1: 288*32*32*16 bf16
#define WS_W2P   (44859392 + 9437184)                // packed W2: 16*8*32*16 bf16
#define WS_X1    (44859392 + 9437184 + 131072)       // intermediate [pixel][512] bf16

// ------------------------------------------------------------------
// fmap: NCHW f32 -> NHWC bf16 (channel dim contiguous for WMMA A frags)
// total = 16*1024*37*37 = 22,429,696 = 87616 * 256
__global__ void k_cvt_fmap(const float* __restrict__ src, bf16_t* __restrict__ dst) {
    int idx = blockIdx.x * 256 + threadIdx.x;
    int x = idx % W_IMG;
    int t = idx / W_IMG;
    int y = t % H_IMG;  t /= H_IMG;
    int c = t % CIN;
    int b = t / CIN;
    dst[(((b * H_IMG + y) * W_IMG + x) << 10) + c] = (bf16_t)src[idx];
}

// ------------------------------------------------------------------
// Pack W1 (OIHW f32) into per-lane WMMA B-fragment layout (bf16):
//   dst[kc][nt][lane][i],  kc in [0,288), nt in [0,32), lane in [0,32), i in [0,16)
//   K = kc*32 + 8*(lane>>4) + (i&7) + 16*(i>>3),  k = (ky*3+kx)*1024 + ci,  col = nt*16 + (lane&15)
__global__ void k_pack_w1(const float* __restrict__ w1, bf16_t* __restrict__ dst) {
    int idx = blockIdx.x * 256 + threadIdx.x;
    int i  = idx & 15;
    int l  = (idx >> 4) & 31;
    int nt = (idx >> 9) & 31;
    int kc = idx >> 14;
    int k  = kc * 32 + ((l >> 4) << 3) + (i & 7) + ((i >> 3) << 4);
    int co = nt * 16 + (l & 15);
    int kk = k >> 10;
    int ci = k & 1023;
    int ky = kk / 3, kx = kk - ky * 3;
    dst[idx] = (bf16_t)w1[((co * CIN + ci) * 3 + ky) * 3 + kx];
}

// ------------------------------------------------------------------
// Pack W2 ([120][512] f32) likewise, N padded to 128 with zeros.
__global__ void k_pack_w2(const float* __restrict__ w2, bf16_t* __restrict__ dst) {
    int idx = blockIdx.x * 256 + threadIdx.x;
    int i  = idx & 15;
    int l  = (idx >> 4) & 31;
    int nt = (idx >> 9) & 7;
    int kc = idx >> 12;
    int k  = kc * 32 + ((l >> 4) << 3) + (i & 7) + ((i >> 3) << 4);
    int co = nt * 16 + (l & 15);
    dst[idx] = (co < COUT) ? (bf16_t)w2[co * CMID + k] : (bf16_t)0.0f;
}

static __device__ __forceinline__ v16bf make_a(const bf16_t* p) {
    v8bf lo = *(const v8bf*)(p);
    v8bf hi = *(const v8bf*)(p + 16);
    v16bf a;
    #pragma unroll
    for (int i = 0; i < 8; ++i) { a[i] = lo[i]; a[i + 8] = hi[i]; }
    return a;
}

// ------------------------------------------------------------------
// conv3x3 + bias + ReLU6 as implicit GEMM: M=21904, N=512, K=9216.
// Block = 256 threads = 8 waves covering 4 M-tiles (64 pixels) x full N=512.
// Each wave: 4(M) x 4(N) accumulators -> 16 WMMAs per k-chunk from
// 4 A frags (LDS) + 4 B frags (global): 0.75 mem-ops per WMMA.
// LDS slice per tap: 64 pixels x 1024 ci bf16 = 128 KB (dynamic LDS).
__global__ __launch_bounds__(256, 1)
void k_conv3x3(const bf16_t* __restrict__ fmapB, const bf16_t* __restrict__ w1p,
               const float* __restrict__ b1, bf16_t* __restrict__ x1) {
    extern __shared__ bf16_t lds[];    // 64 * 1024 bf16 = 128 KB

    int tid  = threadIdx.x;
    int lane = tid & 31;
    int wave = tid >> 5;
    int pixBase = blockIdx.x * 64;

    // staging coords: thread fills row (tid>>2), 256-elem segment (tid&3)
    int row = tid >> 2;                // 0..63
    int seg = tid & 3;
    int pS  = pixBase + row;
    bool rowValid = pS < MTOT;
    int pc   = rowValid ? pS : 0;
    int bImg = pc / NPIX;
    int rem  = pc - bImg * NPIX;
    int y    = rem / W_IMG;
    int x    = rem - y * W_IMG;

    int mRow  = lane & 15;
    int kHalf = lane >> 4;

    v8f acc[4][4] = {};   // [M-tile t][n-tile j], n-tile = wave*4 + j

    for (int kk = 0; kk < 9; ++kk) {
        int ky = kk / 3, kx = kk - (kk / 3) * 3;
        int yy = y + ky - 1, xx = x + kx - 1;
        bool valid = rowValid && ((unsigned)yy < H_IMG) && ((unsigned)xx < W_IMG);

        __syncthreads();
        {
            v8bf* dstL = (v8bf*)(lds + (row << 10) + (seg << 8));
            if (valid) {
                const v8bf* s = (const v8bf*)(fmapB + (((bImg * H_IMG + yy) * W_IMG + xx) << 10)
                                                    + (seg << 8));
                #pragma unroll
                for (int j = 0; j < 16; ++j) dstL[j] = s[j];
            } else {
                v8bf z = {};
                #pragma unroll
                for (int j = 0; j < 16; ++j) dstL[j] = z;
            }
        }
        __syncthreads();

        const bf16_t* aRow0 = lds + ((0 * 16 + mRow) << 10) + kHalf * 8;
        const bf16_t* aRow1 = lds + ((1 * 16 + mRow) << 10) + kHalf * 8;
        const bf16_t* aRow2 = lds + ((2 * 16 + mRow) << 10) + kHalf * 8;
        const bf16_t* aRow3 = lds + ((3 * 16 + mRow) << 10) + kHalf * 8;

        #pragma unroll 1
        for (int c0 = 0; c0 < CIN; c0 += 32) {
            v16bf a0 = make_a(aRow0 + c0);
            v16bf a1 = make_a(aRow1 + c0);
            v16bf a2 = make_a(aRow2 + c0);
            v16bf a3 = make_a(aRow3 + c0);

            int kc = kk * 32 + (c0 >> 5);
            const v16bf* bBase =
                (const v16bf*)(w1p + ((((kc << 5) + (wave << 2)) << 5) + lane) * 16);

            v16bf bf0 = bBase[0 << 5];
            v16bf bf1 = bBase[1 << 5];
            v16bf bf2 = bBase[2 << 5];
            v16bf bf3 = bBase[3 << 5];

            acc[0][0] = __builtin_amdgcn_wmma_f32_16x16x32_bf16(false, a0, false, bf0, (short)0, acc[0][0], false, false);
            acc[1][0] = __builtin_amdgcn_wmma_f32_16x16x32_bf16(false, a1, false, bf0, (short)0, acc[1][0], false, false);
            acc[2][0] = __builtin_amdgcn_wmma_f32_16x16x32_bf16(false, a2, false, bf0, (short)0, acc[2][0], false, false);
            acc[3][0] = __builtin_amdgcn_wmma_f32_16x16x32_bf16(false, a3, false, bf0, (short)0, acc[3][0], false, false);
            acc[0][1] = __builtin_amdgcn_wmma_f32_16x16x32_bf16(false, a0, false, bf1, (short)0, acc[0][1], false, false);
            acc[1][1] = __builtin_amdgcn_wmma_f32_16x16x32_bf16(false, a1, false, bf1, (short)0, acc[1][1], false, false);
            acc[2][1] = __builtin_amdgcn_wmma_f32_16x16x32_bf16(false, a2, false, bf1, (short)0, acc[2][1], false, false);
            acc[3][1] = __builtin_amdgcn_wmma_f32_16x16x32_bf16(false, a3, false, bf1, (short)0, acc[3][1], false, false);
            acc[0][2] = __builtin_amdgcn_wmma_f32_16x16x32_bf16(false, a0, false, bf2, (short)0, acc[0][2], false, false);
            acc[1][2] = __builtin_amdgcn_wmma_f32_16x16x32_bf16(false, a1, false, bf2, (short)0, acc[1][2], false, false);
            acc[2][2] = __builtin_amdgcn_wmma_f32_16x16x32_bf16(false, a2, false, bf2, (short)0, acc[2][2], false, false);
            acc[3][2] = __builtin_amdgcn_wmma_f32_16x16x32_bf16(false, a3, false, bf2, (short)0, acc[3][2], false, false);
            acc[0][3] = __builtin_amdgcn_wmma_f32_16x16x32_bf16(false, a0, false, bf3, (short)0, acc[0][3], false, false);
            acc[1][3] = __builtin_amdgcn_wmma_f32_16x16x32_bf16(false, a1, false, bf3, (short)0, acc[1][3], false, false);
            acc[2][3] = __builtin_amdgcn_wmma_f32_16x16x32_bf16(false, a2, false, bf3, (short)0, acc[2][3], false, false);
            acc[3][3] = __builtin_amdgcn_wmma_f32_16x16x32_bf16(false, a3, false, bf3, (short)0, acc[3][3], false, false);
        }
    }

    // Epilogue: bias + ReLU6, store bf16 intermediate [pixel][512]
    #pragma unroll
    for (int j = 0; j < 4; ++j) {
        int co = ((wave << 2) + j) * 16 + (lane & 15);
        float bias = b1[co];
        #pragma unroll
        for (int t = 0; t < 4; ++t) {
            #pragma unroll
            for (int r = 0; r < 8; ++r) {
                int m = r + ((lane >> 4) << 3);
                int p = pixBase + t * 16 + m;
                float v = fminf(fmaxf(acc[t][j][r] + bias, 0.0f), 6.0f);
                if (p < MTOT) x1[p * CMID + co] = (bf16_t)v;
            }
        }
    }
}

// ------------------------------------------------------------------
// conv1x1 + bias as GEMM: M=21904, N=120 (padded 128), K=512.
// Block = 8 waves x (2 M-tiles each), wave w = N-tile w; B frags reused
// across both M-tiles; output written directly in NHWC = (b,h,w,A,6) flat.
__global__ __launch_bounds__(256, 4)
void k_conv1x1(const bf16_t* __restrict__ x1, const bf16_t* __restrict__ w2p,
               const float* __restrict__ b2, float* __restrict__ out) {
    int tid   = threadIdx.x;
    int lane  = tid & 31;
    int nTile = tid >> 5;
    int pixBase = blockIdx.x * 32;

    int mRow  = lane & 15;
    int kHalf = lane >> 4;
    int pA0 = pixBase + mRow;
    int pA1 = pixBase + 16 + mRow;
    const bf16_t* aBase0 = x1 + (pA0 < MTOT ? pA0 : 0) * CMID + kHalf * 8;
    const bf16_t* aBase1 = x1 + (pA1 < MTOT ? pA1 : 0) * CMID + kHalf * 8;

    v8f acc0 = {}, acc1 = {};
    #pragma unroll
    for (int kc = 0; kc < 16; ++kc) {
        int c0 = kc * 32;
        v16bf a0 = make_a(aBase0 + c0);
        v16bf a1 = make_a(aBase1 + c0);
        v16bf bfrag = *(const v16bf*)(w2p + ((((kc << 3) + nTile) << 5) + lane) * 16);

        acc0 = __builtin_amdgcn_wmma_f32_16x16x32_bf16(false, a0, false, bfrag, (short)0, acc0, false, false);
        acc1 = __builtin_amdgcn_wmma_f32_16x16x32_bf16(false, a1, false, bfrag, (short)0, acc1, false, false);
    }

    int co = nTile * 16 + (lane & 15);
    if (co < COUT) {
        float bias = b2[co];
        #pragma unroll
        for (int r = 0; r < 8; ++r) {
            int m = r + ((lane >> 4) << 3);
            int p0 = pixBase + m;
            int p1 = pixBase + 16 + m;
            if (p0 < MTOT) out[p0 * COUT + co] = acc0[r] + bias;
            if (p1 < MTOT) out[p1 * COUT + co] = acc1[r] + bias;
        }
    }
}

// ------------------------------------------------------------------
extern "C" void kernel_launch(void* const* d_in, const int* in_sizes, int n_in,
                              void* d_out, int out_size, void* d_ws, size_t ws_size,
                              hipStream_t stream) {
    const float* fmap = (const float*)d_in[0];
    const float* W1   = (const float*)d_in[1];
    const float* b1   = (const float*)d_in[2];
    const float* W2   = (const float*)d_in[3];
    const float* b2   = (const float*)d_in[4];
    float* out = (float*)d_out;

    char* ws = (char*)d_ws;
    bf16_t* fmapB = (bf16_t*)(ws + WS_FMAPB);
    bf16_t* w1p   = (bf16_t*)(ws + WS_W1P);
    bf16_t* w2p   = (bf16_t*)(ws + WS_W2P);
    bf16_t* x1    = (bf16_t*)(ws + WS_X1);

    k_cvt_fmap<<<87616, 256, 0, stream>>>(fmap, fmapB);
    k_pack_w1 <<<18432, 256, 0, stream>>>(W1, w1p);
    k_pack_w2 <<<256,   256, 0, stream>>>(W2, w2p);
    k_conv3x3 <<<MBLK4, 256, 131072, stream>>>(fmapB, w1p, b1, x1);
    k_conv1x1 <<<MBLK2, 256, 0, stream>>>(x1, w2p, b2, out);
}
